// Transformer_73993696576038
// MI455X (gfx1250) — matrix operand order repaired and checked by reading, not compiled, not run on hardware
//
#include <hip/hip_runtime.h>
#include <hip/hip_bf16.h>
#include <stdint.h>

// ---------------- model dims ----------------
#define LL   6
#define DD   512
#define HH   8
#define DHD  64
#define DFF  2048
#define SS   50
#define BB   512
#define FF   24
#define OUTN 24
#define MTOK (BB * SS)   // 25600 tokens

typedef __attribute__((ext_vector_type(16))) _Float16 v16h;
typedef __attribute__((ext_vector_type(8)))  _Float16 v8h;
typedef __attribute__((ext_vector_type(8)))  float    v8f;

// =====================================================================
// f32 -> f16 conversion (weights, once per launch)
// =====================================================================
__global__ void cvt_f16_kernel(const float* __restrict__ src,
                               _Float16* __restrict__ dst, int n) {
    int i = blockIdx.x * blockDim.x + threadIdx.x;
    if (i < n) dst[i] = (_Float16)src[i];
}

// =====================================================================
// Encoder: h = x @ Wenc + benc + pe  ;  writes f32 master + f16 shadow
// =====================================================================
__global__ void encode_kernel(const float* __restrict__ x,
                              const float* __restrict__ pe,
                              const float* __restrict__ Wenc,
                              const float* __restrict__ benc,
                              float* __restrict__ H32,
                              _Float16* __restrict__ H16) {
    int idx = blockIdx.x * blockDim.x + threadIdx.x;   // token*512 + d
    int tok = idx >> 9;
    int d   = idx & 511;
    int s   = tok % SS;
    const float* xr = x + (size_t)tok * FF;
    float acc = benc[d] + pe[s * DD + d];
#pragma unroll
    for (int f = 0; f < FF; ++f) acc += xr[f] * Wenc[f * DD + d];
    H32[idx] = acc;
    H16[idx] = (_Float16)acc;
}

// =====================================================================
// WMMA GEMM:  C[M,N] = A[M,K] @ W[K,N] + bias  (compile-time epilogue)
// block tile 128x128, 8 wave32s; each wave 32x64 = 2x4 wmma tiles.
// Register-prefetch pipeline over K (global loads of tile k+1 overlap
// the WMMA work on tile k); L2 prefetch one further tile ahead.
// =====================================================================
#define TM 128
#define TN 128
#define TK 32

template<int STORE32, int STORE16, int RELU>
__global__ __launch_bounds__(256)
void gemm_wmma_kernel(const _Float16* __restrict__ A,
                      const _Float16* __restrict__ W,
                      const float* __restrict__ bias,
                      float* __restrict__ C32,
                      _Float16* __restrict__ C16,
                      int M, int N, int K) {
    __shared__ _Float16 lA[TM][TK];       // 8 KB, row stride 64B (16B aligned)
    __shared__ _Float16 lBt[TN][TK];      // 8 KB, transposed W tile [n][k]

    const int t    = threadIdx.x;
    const int lane = t & 31;
    const int wid  = t >> 5;              // 0..7
    const int wm   = (wid & 3) * 32;      // wave row base in block tile
    const int wn   = (wid >> 2) * 64;     // wave col base in block tile
    const int lh   = lane >> 4;           // K-half selector
    const int l15  = lane & 15;

    const int blockM = blockIdx.y * TM;
    const int blockN = blockIdx.x * TN;
    const int krowA  = K >> 1;            // A row stride in u32
    const int nrowW  = N >> 1;            // W row stride in u32

    const uint32_t* gA = (const uint32_t*)(A + (size_t)blockM * K);
    const uint32_t* gW = (const uint32_t*)(W + blockN);

    // per-thread staging coordinates (fixed): 2048 u32 per tile, 8/thread
    uint32_t ra[8], rw[8];

    auto fetch = [&](int kt) {
        const int ku = kt >> 1;   // u32 column offset into A rows
#pragma unroll
        for (int i = 0; i < 8; ++i) {
            int idx = t + i * 256;
            ra[i] = gA[(size_t)(idx >> 4) * krowA + ku + (idx & 15)];
            rw[i] = gW[(size_t)((idx >> 6) + kt) * nrowW + (idx & 63)];
        }
    };
    auto stage = [&]() {
        uint32_t* sA = (uint32_t*)&lA[0][0];
#pragma unroll
        for (int i = 0; i < 8; ++i) sA[t + i * 256] = ra[i];
#pragma unroll
        for (int i = 0; i < 8; ++i) {
            int idx = t + i * 256;
            int r   = idx >> 6;           // k row 0..31
            int c4  = idx & 63;           // u32 col (2 halves)
            union { uint32_t u; _Float16 h[2]; } cv;
            cv.u = rw[i];
            lBt[2 * c4 + 0][r] = cv.h[0];
            lBt[2 * c4 + 1][r] = cv.h[1];
        }
    };

    v8f c[2][4] = {};
    fetch(0);

    for (int kt = 0; kt < K; kt += TK) {
        stage();                          // prior iteration synced below
        __syncthreads();
        if (kt + TK < K) {
            fetch(kt + TK);               // overlap HBM with WMMA below
            if (kt + 2 * TK < K) {        // warm L2 one tile further out
                __builtin_prefetch(&gA[(size_t)(t >> 4) * krowA + ((kt + 2 * TK) >> 1) + (t & 15)], 0, 1);
                __builtin_prefetch(&gW[(size_t)((t >> 6) + kt + 2 * TK) * nrowW + (t & 63)], 0, 1);
            }
        }

        // ---- fragments per ISA register layouts ----
        v16h a[2], b[4];
#pragma unroll
        for (int tm = 0; tm < 2; ++tm) {
            int row = wm + tm * 16 + l15;
            v8h lo = *(const v8h*)&lA[row][8 * lh];        // K = 8*lh .. +7
            v8h hi = *(const v8h*)&lA[row][16 + 8 * lh];   // K = 16+8*lh .. +7
#pragma unroll
            for (int j = 0; j < 8; ++j) { a[tm][j] = lo[j]; a[tm][j + 8] = hi[j]; }
        }
#pragma unroll
        for (int tn = 0; tn < 4; ++tn) {
            int col = wn + tn * 16 + l15;
            v8h lo = *(const v8h*)&lBt[col][8 * lh];       // K = 8*lh .. +7
            v8h hi = *(const v8h*)&lBt[col][16 + 8 * lh];  // K = 16+8*lh .. +7
#pragma unroll
            for (int j = 0; j < 8; ++j) { b[tn][j] = lo[j]; b[tn][j + 8] = hi[j]; }
        }

        // ---- 2x4 WMMA ----
#pragma unroll
        for (int tm = 0; tm < 2; ++tm)
#pragma unroll
            for (int tn = 0; tn < 4; ++tn)
                c[tm][tn] = __builtin_amdgcn_wmma_f32_16x16x32_f16(
                    false, a[tm], false, b[tn], (short)0, c[tm][tn], false, false);

        __syncthreads();                  // LDS reads done before next stage()
    }

    // ---- branch-free epilogue (template-selected outputs) ----
#pragma unroll
    for (int tm = 0; tm < 2; ++tm) {
#pragma unroll
        for (int tn = 0; tn < 4; ++tn) {
            int col = blockN + wn + tn * 16 + l15;
            float bv = bias[col];
            int rowb = blockM + wm + tm * 16 + 8 * lh;
            size_t ob = (size_t)rowb * N + col;
#pragma unroll
            for (int r = 0; r < 8; ++r) {
                float v = c[tm][tn][r] + bv;
                if (RELU) v = v > 0.0f ? v : 0.0f;
                size_t o = ob + (size_t)r * N;
                if (STORE32) C32[o] = v;
                if (STORE16) C16[o] = (_Float16)v;
            }
        }
    }
}

// =====================================================================
// Attention: per (b, h) block; S=50, dh=64; scalar VALU (<2% of FLOPs)
// =====================================================================
__global__ __launch_bounds__(64)
void attention_kernel(const _Float16* __restrict__ Q,
                      const _Float16* __restrict__ K,
                      const _Float16* __restrict__ V,
                      _Float16* __restrict__ O) {
    __shared__ float sQ[SS][DHD];
    __shared__ float sK[SS][DHD];
    __shared__ float sV[SS][DHD];
    __shared__ float sS[64][SS + 2];

    int b = blockIdx.x >> 3;     // / H
    int h = blockIdx.x & 7;      // % H
    int t = threadIdx.x;
    size_t base = (size_t)b * SS * DD + (size_t)h * DHD;

    for (int i = t; i < SS * DHD; i += 64) {
        int r = i >> 6, c = i & 63;
        size_t g = base + (size_t)r * DD + c;
        sQ[r][c] = (float)Q[g];
        sK[r][c] = (float)K[g];
        sV[r][c] = (float)V[g];
    }
    __syncthreads();

    if (t < SS) {
        float mx = -1e30f;
        for (int kj = 0; kj < SS; ++kj) {
            float acc = 0.0f;
#pragma unroll 8
            for (int d = 0; d < DHD; ++d) acc += sQ[t][d] * sK[kj][d];
            acc *= 0.125f;                 // 1/sqrt(64)
            sS[t][kj] = acc;
            mx = fmaxf(mx, acc);
        }
        float sum = 0.0f;
        for (int kj = 0; kj < SS; ++kj) {
            float e = __expf(sS[t][kj] - mx);
            sS[t][kj] = e;
            sum += e;
        }
        float inv = 1.0f / sum;
        for (int d = 0; d < DHD; ++d) {
            float acc = 0.0f;
#pragma unroll 5
            for (int kj = 0; kj < SS; ++kj) acc += sS[t][kj] * sV[kj][d];
            O[base + (size_t)t * DD + d] = (_Float16)(acc * inv);
        }
    }
}

// =====================================================================
// Fused residual add + LayerNorm: H = LN(X + Dl)*g + be ; f32 + f16 out
// =====================================================================
__global__ __launch_bounds__(256)
void add_ln_kernel(const float* __restrict__ X, const float* __restrict__ Dl,
                   const float* __restrict__ g, const float* __restrict__ be,
                   float* __restrict__ H32, _Float16* __restrict__ H16) {
    __shared__ float red[256];
    int row = blockIdx.x;
    int t = threadIdx.x;
    const float* xr = X  + (size_t)row * DD;
    const float* dr = Dl + (size_t)row * DD;
    float v0 = xr[t] + dr[t];
    float v1 = xr[t + 256] + dr[t + 256];

    red[t] = v0 + v1;
    __syncthreads();
    for (int s = 128; s > 0; s >>= 1) {
        if (t < s) red[t] += red[t + s];
        __syncthreads();
    }
    float mean = red[0] * (1.0f / 512.0f);
    __syncthreads();

    float d0 = v0 - mean, d1 = v1 - mean;
    red[t] = d0 * d0 + d1 * d1;
    __syncthreads();
    for (int s = 128; s > 0; s >>= 1) {
        if (t < s) red[t] += red[t + s];
        __syncthreads();
    }
    float rstd = rsqrtf(red[0] * (1.0f / 512.0f) + 1e-6f);

    float o0 = d0 * rstd * g[t]       + be[t];
    float o1 = d1 * rstd * g[t + 256] + be[t + 256];
    size_t o = (size_t)row * DD + t;
    H32[o]        = o0;  H16[o]        = (_Float16)o0;
    H32[o + 256]  = o1;  H16[o + 256]  = (_Float16)o1;
}

// =====================================================================
// Output head: out = sigmoid(h @ Wout + bout)
// =====================================================================
__global__ void out_kernel(const float* __restrict__ Hf,
                           const float* __restrict__ Wout,
                           const float* __restrict__ bout,
                           float* __restrict__ out) {
    int idx = blockIdx.x * blockDim.x + threadIdx.x;
    if (idx >= MTOK * OUTN) return;
    int tok = idx / OUTN;
    int o   = idx % OUTN;
    const float* hr = Hf + (size_t)tok * DD;
    float acc = bout[o];
#pragma unroll 8
    for (int d = 0; d < DD; ++d) acc += hr[d] * Wout[d * OUTN + o];
    out[idx] = 1.0f / (1.0f + __expf(-acc));
}

// =====================================================================
// host driver
// =====================================================================
extern "C" void kernel_launch(void* const* d_in, const int* in_sizes, int n_in,
                              void* d_out, int out_size, void* d_ws, size_t ws_size,
                              hipStream_t stream) {
    (void)in_sizes; (void)n_in; (void)out_size; (void)ws_size;

    const float* x    = (const float*)d_in[0];
    const float* pe   = (const float*)d_in[1];
    const float* Wenc = (const float*)d_in[2];
    const float* benc = (const float*)d_in[3];
    const float* Wq   = (const float*)d_in[4];
    const float* bq   = (const float*)d_in[5];
    const float* Wk   = (const float*)d_in[6];
    const float* bk   = (const float*)d_in[7];
    const float* Wv   = (const float*)d_in[8];
    const float* bv   = (const float*)d_in[9];
    const float* Wo   = (const float*)d_in[10];
    const float* bo   = (const float*)d_in[11];
    const float* W1   = (const float*)d_in[12];
    const float* b1   = (const float*)d_in[13];
    const float* W2   = (const float*)d_in[14];
    const float* b2   = (const float*)d_in[15];
    const float* g1   = (const float*)d_in[16];
    const float* be1  = (const float*)d_in[17];
    const float* g2   = (const float*)d_in[18];
    const float* be2  = (const float*)d_in[19];
    const float* Wout = (const float*)d_in[20];
    const float* bout = (const float*)d_in[21];
    float* out = (float*)d_out;

    // ---- carve workspace ----
    char* ws = (char*)d_ws;
    auto alloc = [&](size_t bytes) -> char* {
        char* p = ws;
        ws += (bytes + 255) & ~(size_t)255;
        return p;
    };
    const size_t MD  = (size_t)MTOK * DD;
    const size_t MF  = (size_t)MTOK * DFF;
    float*    h32    = (float*)   alloc(MD * 4);
    float*    tmp32  = (float*)   alloc(MD * 4);
    _Float16* h16    = (_Float16*)alloc(MD * 2);
    _Float16* q16    = (_Float16*)alloc(MD * 2);
    _Float16* k16    = (_Float16*)alloc(MD * 2);
    _Float16* v16    = (_Float16*)alloc(MD * 2);
    _Float16* o16    = (_Float16*)alloc(MD * 2);
    _Float16* f16b   = (_Float16*)alloc(MF * 2);
    _Float16* Wq16   = (_Float16*)alloc((size_t)LL * DD * DD * 2);
    _Float16* Wk16   = (_Float16*)alloc((size_t)LL * DD * DD * 2);
    _Float16* Wv16   = (_Float16*)alloc((size_t)LL * DD * DD * 2);
    _Float16* Wo16   = (_Float16*)alloc((size_t)LL * DD * DD * 2);
    _Float16* W116   = (_Float16*)alloc((size_t)LL * DD * DFF * 2);
    _Float16* W216   = (_Float16*)alloc((size_t)LL * DFF * DD * 2);

    // ---- weight conversion f32 -> f16 ----
    {
        int nW = LL * DD * DD;
        int nB = (nW + 255) / 256;
        cvt_f16_kernel<<<nB, 256, 0, stream>>>(Wq, Wq16, nW);
        cvt_f16_kernel<<<nB, 256, 0, stream>>>(Wk, Wk16, nW);
        cvt_f16_kernel<<<nB, 256, 0, stream>>>(Wv, Wv16, nW);
        cvt_f16_kernel<<<nB, 256, 0, stream>>>(Wo, Wo16, nW);
        int nF = LL * DD * DFF;
        int nBf = (nF + 255) / 256;
        cvt_f16_kernel<<<nBf, 256, 0, stream>>>(W1, W116, nF);
        cvt_f16_kernel<<<nBf, 256, 0, stream>>>(W2, W216, nF);
    }

    // ---- encoder ----
    encode_kernel<<<(MTOK * DD) / 256, 256, 0, stream>>>(x, pe, Wenc, benc, h32, h16);

    const dim3 blk(256);
    const dim3 gD (DD  / TN, MTOK / TM);   // (4, 200)
    const dim3 gF (DFF / TN, MTOK / TM);   // (16, 200)

    for (int i = 0; i < LL; ++i) {
        const size_t wo  = (size_t)i * DD * DD;
        const size_t wf1 = (size_t)i * DD * DFF;
        const size_t wf2 = (size_t)i * DFF * DD;

        // Q / K / V projections (f16 out for attention)
        gemm_wmma_kernel<0,1,0><<<gD, blk, 0, stream>>>(h16, Wq16 + wo, bq + i * DD,
                                                        nullptr, q16, MTOK, DD, DD);
        gemm_wmma_kernel<0,1,0><<<gD, blk, 0, stream>>>(h16, Wk16 + wo, bk + i * DD,
                                                        nullptr, k16, MTOK, DD, DD);
        gemm_wmma_kernel<0,1,0><<<gD, blk, 0, stream>>>(h16, Wv16 + wo, bv + i * DD,
                                                        nullptr, v16, MTOK, DD, DD);

        // attention per (b, h)
        attention_kernel<<<BB * HH, 64, 0, stream>>>(q16, k16, v16, o16);

        // output projection (f32 out for residual)
        gemm_wmma_kernel<1,0,0><<<gD, blk, 0, stream>>>(o16, Wo16 + wo, bo + i * DD,
                                                        tmp32, nullptr, MTOK, DD, DD);

        // h = LN(h + attn_out)
        add_ln_kernel<<<MTOK, 256, 0, stream>>>(h32, tmp32, g1 + i * DD, be1 + i * DD,
                                                h32, h16);

        // FFN: relu(h @ W1 + b1) @ W2 + b2
        gemm_wmma_kernel<0,1,1><<<gF, blk, 0, stream>>>(h16, W116 + wf1, b1 + i * DFF,
                                                        nullptr, f16b, MTOK, DFF, DD);
        gemm_wmma_kernel<1,0,0><<<gD, blk, 0, stream>>>(f16b, W216 + wf2, b2 + i * DD,
                                                        tmp32, nullptr, MTOK, DD, DFF);

        // h = LN(h + ffn_out)
        add_ln_kernel<<<MTOK, 256, 0, stream>>>(h32, tmp32, g2 + i * DD, be2 + i * DD,
                                                h32, h16);
    }

    // ---- output head ----
    out_kernel<<<(MTOK * OUTN + 255) / 256, 256, 0, stream>>>(h32, Wout, bout, out);
}
